// Model_17824114278658
// MI455X (gfx1250) — matrix operand order repaired and checked by reading, not compiled, not run on hardware
//
#include <hip/hip_runtime.h>
#include <hip/hip_fp16.h>
#include <cstdint>

typedef __attribute__((ext_vector_type(16))) _Float16 v16h;
typedef __attribute__((ext_vector_type(8)))  _Float16 v8h;
typedef __attribute__((ext_vector_type(4)))  _Float16 v4h;
typedef __attribute__((ext_vector_type(8)))  float    v8f;
typedef __attribute__((ext_vector_type(4)))  unsigned int u32x4;
typedef __attribute__((ext_vector_type(8)))  int      i32x8;
typedef __attribute__((ext_vector_type(4)))  int      i32x4;

#define EPSBN 1e-4f
#define NB 2
#define D0C 150
#define H0C 150
#define W0C 300
#define D1C 50
#define H1C 50
#define W1C 100
#define S1C (D1C*H1C*W1C)      /* 250000 */
#define D5C 46
#define H5C 46
#define W5C 96
#define S5C (D5C*H5C*W5C)      /* 203136 */
#define FLATC (32*S5C)         /* 6500352 */
#define LCHUNK 8192

// ---------------- stage 1: submanifold conv(1->4) + sparse maxpool 3/3 ----------------
__global__ void k_in_pool(const float* __restrict__ x, const float* __restrict__ w_in,
                          float* __restrict__ h1, uint8_t* __restrict__ mask1) {
  int idx = blockIdx.x * blockDim.x + threadIdx.x;
  if (idx >= NB * S1C) return;
  int b = idx / S1C, site = idx % S1C;
  int d1 = site / (H1C * W1C); int rem = site % (H1C * W1C);
  int hh1 = rem / W1C, ww1 = rem % W1C;
  const float* xb = x + (size_t)b * D0C * H0C * W0C;
  float ninf = -__builtin_inff();
  float mx0 = ninf, mx1 = ninf, mx2 = ninf, mx3 = ninf;
  bool any = false;
  for (int qd = 0; qd < 3; ++qd)
    for (int qh = 0; qh < 3; ++qh)
      for (int qw = 0; qw < 3; ++qw) {
        int dz = d1 * 3 + qd, dy = hh1 * 3 + qh, dx = ww1 * 3 + qw;
        float xc = xb[((size_t)dz * H0C + dy) * W0C + dx];
        if (xc == 0.f) continue;           // submanifold: only active sites produce output
        any = true;
        float a0 = 0.f, a1 = 0.f, a2 = 0.f, a3 = 0.f;
        for (int nd = -1; nd <= 1; ++nd)
          for (int nh = -1; nh <= 1; ++nh)
            for (int nw = -1; nw <= 1; ++nw) {
              int z = dz + nd, y = dy + nh, xx = dx + nw;
              float v = 0.f;
              if ((unsigned)z < D0C && (unsigned)y < H0C && (unsigned)xx < W0C)
                v = xb[((size_t)z * H0C + y) * W0C + xx];
              int ni = (nd + 1) * 9 + (nh + 1) * 3 + (nw + 1);
              a0 += w_in[ni] * v;  a1 += w_in[27 + ni] * v;
              a2 += w_in[54 + ni] * v;  a3 += w_in[81 + ni] * v;
            }
        mx0 = fmaxf(mx0, a0); mx1 = fmaxf(mx1, a1);
        mx2 = fmaxf(mx2, a2); mx3 = fmaxf(mx3, a3);
      }
  mask1[idx] = any ? 1 : 0;
  size_t o = (size_t)b * 4 * S1C + site;
  h1[o]           = any ? mx0 : 0.f;
  h1[o + S1C]     = any ? mx1 : 0.f;
  h1[o + 2 * S1C] = any ? mx2 : 0.f;
  h1[o + 3 * S1C] = any ? mx3 : 0.f;
}

// ---------------- BN+ReLU (eval stats), masked, emit fp16 CHANNEL-LAST [b][site][c] ----------------
__global__ void k_bnrelu16cl(const float* __restrict__ h, const float* __restrict__ g,
                             const float* __restrict__ be, const float* __restrict__ mn,
                             const float* __restrict__ vr, const uint8_t* __restrict__ mask,
                             _Float16* __restrict__ y, int C) {
  int idx = blockIdx.x * blockDim.x + threadIdx.x;
  if (idx >= NB * C * S1C) return;
  int c = idx % C;
  int site = (idx / C) % S1C;
  int b = idx / (C * S1C);
  float v = 0.f;
  if (mask[(size_t)b * S1C + site]) {
    float s = g[c] * rsqrtf(vr[c] + EPSBN);
    v = fmaxf((h[((size_t)b * C + c) * S1C + site] - mn[c]) * s + be[c], 0.f);
  }
  y[idx] = (_Float16)v;
}

// identity cast, channel-major f32 -> channel-last f16
__global__ void k_cast16cl(const float* __restrict__ h, _Float16* __restrict__ y, int C) {
  int idx = blockIdx.x * blockDim.x + threadIdx.x;
  if (idx >= NB * C * S1C) return;
  int c = idx % C;
  int site = (idx / C) % S1C;
  int b = idx / (C * S1C);
  y[idx] = (_Float16)h[((size_t)b * C + c) * S1C + site];
}

// ---------------- pack conv weights [Cout][Cin][k^3] -> WMMA B operand layout ----------------
// K ordering is neighbor-major: K = ng*Cin + cin  (matches A gather)
// out linear index i = ((nt*Kb + kb)*32 + lane)*16 + j ; K = kb*32 + (j>=8)*16 + (lane>=16)*8 + (j&7) ; N = nt*16 + (lane&15)
__global__ void k_pack(const float* __restrict__ w, _Float16* __restrict__ out,
                       int Cin, int kcube, int Kpad, int Cout, int ntiles) {
  int Kreal = Cin * kcube;
  int Kb = Kpad / 32;
  long tot = (long)ntiles * Kb * 32 * 16;
  long i = (long)blockIdx.x * blockDim.x + threadIdx.x;
  if (i >= tot) return;
  int j = (int)(i & 15);
  int lane = (int)((i >> 4) & 31);
  long rest = i >> 9;
  int kb = (int)(rest % Kb);
  int nt = (int)(rest / Kb);
  int K = kb * 32 + ((j >> 3) << 4) + ((lane >> 4) << 3) + (j & 7);
  int n = nt * 16 + (lane & 15);
  float v = 0.f;
  if (K < Kreal && n < Cout) {
    int ng = K / Cin, cin = K % Cin;
    v = w[(size_t)n * Kreal + (size_t)cin * kcube + ng];
  }
  out[i] = (_Float16)v;
}

// ---------------- implicit-GEMM conv via V_WMMA_F32_16X16X32_F16, templated geometry ----------------
#define EPI_MASK 0   // out = mask ? acc : 0
#define EPI_RES  1   // out = mask ? acc + resid[n] : 0
#define EPI_NIN  2   // out = mask ? acc + sum_c nin[n,c]*resid[c] : 0
#define EPI_BN5  3   // out = mask ? relu(bn(acc)) : 0
template<int CIN, int KS>
__global__ void __launch_bounds__(32)
k_conv_wmma(const _Float16* __restrict__ A, const _Float16* __restrict__ Bw,
            const uint8_t* __restrict__ mask, float* __restrict__ out,
            const float* __restrict__ resid, const float* __restrict__ nin, int Cres,
            const float* __restrict__ g, const float* __restrict__ be,
            const float* __restrict__ mn, const float* __restrict__ vr,
            int Cout, int mode) {
  constexpr int PAD   = (KS == 3) ? 1 : 0;
  constexpr int DIN = D1C, HIN = H1C, WIN = W1C;
  constexpr int DOUT = (KS == 3) ? D1C : D5C;
  constexpr int HOUT = (KS == 3) ? H1C : H5C;
  constexpr int WOUT = (KS == 3) ? W1C : W5C;
  constexpr int KCUBE = KS * KS * KS;
  constexpr int KREAL = CIN * KCUBE;
  constexpr int KPAD  = ((KREAL + 31) / 32) * 32;
  constexpr int KB    = KPAD / 32;
  constexpr int SOUT  = DOUT * HOUT * WOUT;

  int lane = threadIdx.x;
  int m = lane & 15, kg = lane >> 4;
  int tile = blockIdx.x, nt = blockIdx.y;
  long row = (long)tile * 16 + m;
  int b = (int)(row / SOUT);
  int site = (int)(row % SOUT);
  int dz = site / (HOUT * WOUT); int rr = site % (HOUT * WOUT);
  int dy = rr / WOUT, dx = rr % WOUT;
  const _Float16* Ab = A + (long)b * ((long)DIN * HIN * WIN) * CIN;  // channel-last

  // --- Stage the whole packed-B N-tile into LDS via the Tensor Data Mover (k3 convs only:
  // tile is Kb*1024 bytes <= 21.5KB, so occupancy is unharmed; k5 tile would be 96KB -> skip).
  __shared__ __align__(32) _Float16 ldsB[(KS == 3) ? (KPAD * 16) : 1];
  if constexpr (KS == 3) {
    unsigned lds_off = (unsigned)(unsigned long long)(uintptr_t)&ldsB[0];  // LDS aperture: low 32 bits = LDS offset
    unsigned long long ga = (unsigned long long)(uintptr_t)(Bw + (size_t)nt * KB * 512);
    unsigned nE = (unsigned)(KB * 128);        // tile length in 8-byte elements
    // D# group0: count=1 | lds_addr[63:32] | global_addr[120:64] | type=2 [127:126]
    u32x4 g0;
    g0[0] = 1u;
    g0[1] = lds_off;
    g0[2] = (unsigned)(ga & 0xFFFFFFFFu);
    g0[3] = (unsigned)((ga >> 32) & 0x01FFFFFFu) | (2u << 30);
    // D# group1: wg_mask=0, data_size=3 (8B) [17:16], tensor_dim0 [79:48]=nE,
    // tensor_dim1 [111:80]=1, tile_dim0 [127:112]=nE, tile_dim1 [143:128]=1, stride0 [207:160]=nE
    i32x8 g1;
    g1[0] = (int)(3u << 16);
    g1[1] = (int)((nE & 0xFFFFu) << 16);
    g1[2] = (int)((nE >> 16) | (1u << 16));
    g1[3] = (int)((nE & 0xFFFFu) << 16);
    g1[4] = 1;
    g1[5] = (int)nE;
    g1[6] = 0;
    g1[7] = 0;
    i32x4 gz = {0, 0, 0, 0};
    i32x8 gz8 = {0, 0, 0, 0, 0, 0, 0, 0};
    __builtin_amdgcn_tensor_load_to_lds(g0, g1, gz, gz, gz8, 0);
    __builtin_amdgcn_s_wait_tensorcnt(0);
    asm volatile("" ::: "memory");
  }

  v8f acc = {0.f, 0.f, 0.f, 0.f, 0.f, 0.f, 0.f, 0.f};

  auto step = [&](int kb) {
    v16h a;
    #pragma unroll
    for (int grp = 0; grp < 2; ++grp) {
      int K0 = kb * 32 + kg * 8 + grp * 16;       // ISA A 16x32 per-lane K map (8-aligned groups)
      if constexpr (CIN >= 8) {
        int ng = K0 / CIN, c0 = K0 % CIN;         // compile-time CIN: folds to shifts/mads
        v8h val = {(_Float16)0.f,(_Float16)0.f,(_Float16)0.f,(_Float16)0.f,
                   (_Float16)0.f,(_Float16)0.f,(_Float16)0.f,(_Float16)0.f};
        if (ng < KCUBE) {
          int nd = ng / (KS * KS), r2 = ng % (KS * KS);
          int zi = dz + nd - PAD, yi = dy + r2 / KS - PAD, xi = dx + r2 % KS - PAD;
          if ((unsigned)zi < (unsigned)DIN && (unsigned)yi < (unsigned)HIN && (unsigned)xi < (unsigned)WIN)
            val = *(const v8h*)(Ab + ((long)(zi * HIN + yi) * WIN + xi) * CIN + c0);  // global_load_b128
        }
        #pragma unroll
        for (int j = 0; j < 8; ++j) a[grp * 8 + j] = val[j];
      } else {  // CIN == 4: 8-group spans two neighbors, two b64 loads
        #pragma unroll
        for (int h4 = 0; h4 < 2; ++h4) {
          int ng = K0 / CIN + h4;
          v4h val = {(_Float16)0.f,(_Float16)0.f,(_Float16)0.f,(_Float16)0.f};
          if (ng < KCUBE) {
            int nd = ng / (KS * KS), r2 = ng % (KS * KS);
            int zi = dz + nd - PAD, yi = dy + r2 / KS - PAD, xi = dx + r2 % KS - PAD;
            if ((unsigned)zi < (unsigned)DIN && (unsigned)yi < (unsigned)HIN && (unsigned)xi < (unsigned)WIN)
              val = *(const v4h*)(Ab + ((long)(zi * HIN + yi) * WIN + xi) * CIN);
          }
          #pragma unroll
          for (int j = 0; j < 4; ++j) a[grp * 8 + h4 * 4 + j] = val[j];
        }
      }
    }
    v16h bv;
    if constexpr (KS == 3) {
      bv = *(const v16h*)(&ldsB[(size_t)kb * 512 + lane * 16]);   // ds_load_b128 x2 from TDM-staged tile
    } else {
      bv = *(const v16h*)(Bw + (((long)nt * KB + kb) * 32 + lane) * 16);
    }
    acc = __builtin_amdgcn_wmma_f32_16x16x32_f16(false, a, false, bv, (short)0, acc, false, false);
  };
  if constexpr (KS == 3) {
    #pragma unroll
    for (int kb = 0; kb < KB; ++kb) step(kb);
  } else {
    #pragma unroll 2
    for (int kb = 0; kb < KB; ++kb) step(kb);
  }

  int n = nt * 16 + m;
  if (n < Cout) {
    #pragma unroll
    for (int r8 = 0; r8 < 8; ++r8) {
      int M = r8 + kg * 8;                               // ISA C/D map
      long orow = (long)tile * 16 + M;
      int ob = (int)(orow / SOUT);
      int osite = (int)(orow % SOUT);
      float o = 0.f;
      if (mask[(size_t)ob * SOUT + osite]) {
        float val = acc[r8];
        if (mode == EPI_BN5) {
          float s = g[n] * rsqrtf(vr[n] + EPSBN);
          o = fmaxf((val - mn[n]) * s + be[n], 0.f);
        } else {
          o = val;
          if (mode == EPI_RES) {
            o += resid[((size_t)ob * Cres + n) * SOUT + osite];
          } else if (mode == EPI_NIN) {
            float s2 = 0.f;
            for (int c = 0; c < Cres; ++c)
              s2 += nin[n * Cres + c] * resid[((size_t)ob * Cres + c) * SOUT + osite];
            o += s2;
          }
        }
      }
      out[((size_t)ob * Cout + n) * SOUT + osite] = o;
    }
  }
}

// ---------------- m5: any active in 5^3 window (VALID) ----------------
__global__ void k_m5(const uint8_t* __restrict__ mask1, uint8_t* __restrict__ m5) {
  int idx = blockIdx.x * blockDim.x + threadIdx.x;
  if (idx >= NB * S5C) return;
  int b = idx / S5C, site = idx % S5C;
  int d = site / (H5C * W5C); int rem = site % (H5C * W5C);
  int h = rem / W5C, w = rem % W5C;
  const uint8_t* mb = mask1 + (size_t)b * S1C;
  uint8_t any = 0;
  for (int i = 0; i < 5 && !any; ++i)
    for (int j = 0; j < 5 && !any; ++j)
      for (int k = 0; k < 5; ++k)
        if (mb[((size_t)(d + i) * H1C + (h + j)) * W1C + (w + k)]) { any = 1; break; }
  m5[idx] = any;
}

// ---------------- big GEMV: [2,FLAT] x [FLAT,32] -- HBM-bound, float4 loads ----------------
__global__ void __launch_bounds__(256)
k_lin1(const float* __restrict__ w, const float* __restrict__ h5, float* __restrict__ part) {
  int ch = blockIdx.x, og = blockIdx.y;            // og in 0..3 -> 8 outputs each
  long f0 = (long)ch * LCHUNK;
  long fend = f0 + LCHUNK; if (fend > FLATC) fend = FLATC;
  float acc[16];
  #pragma unroll
  for (int q = 0; q < 16; ++q) acc[q] = 0.f;
  const float4* h4a = (const float4*)h5;
  const float4* h4b = (const float4*)(h5 + FLATC);
  for (long q = f0 / 4 + threadIdx.x; q < fend / 4; q += 256) {
    float4 h0 = h4a[q];
    float4 h1 = h4b[q];
    #pragma unroll
    for (int oo = 0; oo < 8; ++oo) {
      const float4* wr = (const float4*)(w + (size_t)(og * 8 + oo) * FLATC);
      float4 wv = wr[q];
      acc[oo * 2]     += wv.x * h0.x + wv.y * h0.y + wv.z * h0.z + wv.w * h0.w;
      acc[oo * 2 + 1] += wv.x * h1.x + wv.y * h1.y + wv.z * h1.z + wv.w * h1.w;
    }
  }
  __shared__ float red[256];
  for (int q = 0; q < 16; ++q) {
    red[threadIdx.x] = acc[q];
    __syncthreads();
    for (int s = 128; s > 0; s >>= 1) {
      if ((int)threadIdx.x < s) red[threadIdx.x] += red[threadIdx.x + s];
      __syncthreads();
    }
    if (threadIdx.x == 0) part[(size_t)ch * 64 + og * 16 + q] = red[0];
    __syncthreads();
  }
}

__global__ void k_lin1red(const float* __restrict__ part, const float* __restrict__ b1,
                          float* __restrict__ a, int nchunks) {
  int t = threadIdx.x;
  if (t >= 64) return;
  float s = 0.f;
  for (int ch = 0; ch < nchunks; ++ch) s += part[(size_t)ch * 64 + t];
  int og = t >> 4, q = t & 15, oo = q >> 1, b = q & 1, o = og * 8 + oo;
  float v = s + b1[o];
  a[b * 32 + o] = (v > 0.f) ? v : (expf(v) - 1.f);   // ELU
}

__global__ void k_head(const float* __restrict__ a, const float* __restrict__ w2,
                       const float* __restrict__ b2, float* __restrict__ outp) {
  int b = threadIdx.x;
  if (b >= NB) return;
  float lg[3];
  for (int k = 0; k < 3; ++k) {
    float s = b2[k];
    for (int o = 0; o < 32; ++o) s += w2[k * 32 + o] * a[b * 32 + o];
    lg[k] = s;
  }
  float mx = fmaxf(lg[0], fmaxf(lg[1], lg[2]));
  float e0 = expf(lg[0] - mx), e1 = expf(lg[1] - mx), e2 = expf(lg[2] - mx);
  float inv = 1.f / (e0 + e1 + e2);
  outp[b * 3 + 0] = e0 * inv;
  outp[b * 3 + 1] = e1 * inv;
  outp[b * 3 + 2] = e2 * inv;
}

// ---------------- conv dispatch over compile-time (Cin, ks) ----------------
static void launch_conv(int Cin, int ks, dim3 gr, hipStream_t st,
                        const _Float16* A, const _Float16* Bw, const uint8_t* mask, float* out,
                        const float* resid, const float* nin, int Cres,
                        const float* g, const float* be, const float* mn, const float* vr,
                        int Cout, int mode) {
  if (ks == 5) {
    k_conv_wmma<24, 5><<<gr, 32, 0, st>>>(A, Bw, mask, out, resid, nin, Cres, g, be, mn, vr, Cout, mode);
    return;
  }
  switch (Cin) {
    case 4:  k_conv_wmma<4, 3><<<gr, 32, 0, st>>>(A, Bw, mask, out, resid, nin, Cres, g, be, mn, vr, Cout, mode); break;
    case 8:  k_conv_wmma<8, 3><<<gr, 32, 0, st>>>(A, Bw, mask, out, resid, nin, Cres, g, be, mn, vr, Cout, mode); break;
    case 16: k_conv_wmma<16, 3><<<gr, 32, 0, st>>>(A, Bw, mask, out, resid, nin, Cres, g, be, mn, vr, Cout, mode); break;
    default: k_conv_wmma<24, 3><<<gr, 32, 0, st>>>(A, Bw, mask, out, resid, nin, Cres, g, be, mn, vr, Cout, mode); break;
  }
}

// ---------------- host orchestration ----------------
extern "C" void kernel_launch(void* const* d_in, const int* in_sizes, int n_in,
                              void* d_out, int out_size, void* d_ws, size_t ws_size,
                              hipStream_t stream) {
  (void)in_sizes; (void)n_in; (void)out_size; (void)ws_size;
  const float* x    = (const float*)d_in[0];
  const float* w_in = (const float*)d_in[1];
  const float* w5   = (const float*)d_in[65];
  const float* bn5g = (const float*)d_in[66];
  const float* bn5b = (const float*)d_in[67];
  const float* bn5m = (const float*)d_in[68];
  const float* bn5v = (const float*)d_in[69];
  const float* wl1  = (const float*)d_in[70];
  const float* bl1  = (const float*)d_in[71];
  const float* wl2  = (const float*)d_in[72];
  const float* bl2  = (const float*)d_in[73];

  uint8_t* base = (uint8_t*)d_ws;
  size_t off = 0;
  auto alloc = [&](size_t bytes) -> void* {
    off = (off + 255) & ~(size_t)255;
    void* p = base + off;
    off += bytes;
    return p;
  };
  uint8_t*  mask1 = (uint8_t*)alloc((size_t)NB * S1C);
  uint8_t*  m5    = (uint8_t*)alloc((size_t)NB * S5C);
  float*    bufA  = (float*)alloc((size_t)NB * 32 * S5C * sizeof(float));
  float*    bufB  = (float*)alloc((size_t)NB * 32 * S5C * sizeof(float));
  _Float16* y16   = (_Float16*)alloc((size_t)NB * 24 * S1C * sizeof(_Float16));
  _Float16* wpack = (_Float16*)alloc((size_t)2 * 94 * 32 * 16 * sizeof(_Float16));
  int nchunks = (FLATC + LCHUNK - 1) / LCHUNK;
  float* part = (float*)alloc((size_t)nchunks * 64 * sizeof(float));
  float* avec = (float*)alloc(64 * sizeof(float));

  // stage 1
  {
    int n = NB * S1C;
    k_in_pool<<<(n + 255) / 256, 256, 0, stream>>>(x, w_in, bufA, mask1);
  }

  float* hcur = bufA;
  float* hnxt = bufB;
  struct Blk { int ci, co, base, nin; };
  const Blk blks[6] = { {4,8,2,1}, {8,8,13,0}, {8,16,23,1}, {16,16,34,0}, {16,24,44,1}, {24,24,55,0} };
  int mtiles = NB * S1C / 16;   // 31250, batches never straddle a tile

  for (int bi = 0; bi < 6; ++bi) {
    const Blk& B = blks[bi];
    const float* g1 = (const float*)d_in[B.base + 0];
    const float* e1 = (const float*)d_in[B.base + 1];
    const float* m1 = (const float*)d_in[B.base + 2];
    const float* v1 = (const float*)d_in[B.base + 3];
    const float* cw1 = (const float*)d_in[B.base + 4];
    const float* g2 = (const float*)d_in[B.base + 5];
    const float* e2 = (const float*)d_in[B.base + 6];
    const float* m2 = (const float*)d_in[B.base + 7];
    const float* v2 = (const float*)d_in[B.base + 8];
    const float* cw2 = (const float*)d_in[B.base + 9];
    const float* nin = B.nin ? (const float*)d_in[B.base + 10] : nullptr;
    int Kp1 = ((B.ci * 27 + 31) / 32) * 32;
    int Kp2 = ((B.co * 27 + 31) / 32) * 32;
    int nt = (B.co + 15) / 16;
    dim3 gr((unsigned)mtiles, (unsigned)nt);

    // conv1: y = conv(bnrelu(h, bn1)) * mask
    { long tot = (long)nt * (Kp1 / 32) * 512;
      k_pack<<<(unsigned)((tot + 255) / 256), 256, 0, stream>>>(cw1, wpack, B.ci, 27, Kp1, B.co, nt); }
    { int n = NB * B.ci * S1C;
      k_bnrelu16cl<<<(n + 255) / 256, 256, 0, stream>>>(hcur, g1, e1, m1, v1, mask1, y16, B.ci); }
    launch_conv(B.ci, 3, gr, stream, y16, wpack, mask1, hnxt,
                nullptr, nullptr, 0, nullptr, nullptr, nullptr, nullptr, B.co, EPI_MASK);

    // conv2: h = conv(bnrelu(y, bn2)) * mask + residual(h)
    { long tot = (long)nt * (Kp2 / 32) * 512;
      k_pack<<<(unsigned)((tot + 255) / 256), 256, 0, stream>>>(cw2, wpack, B.co, 27, Kp2, B.co, nt); }
    { int n = NB * B.co * S1C;
      k_bnrelu16cl<<<(n + 255) / 256, 256, 0, stream>>>(hnxt, g2, e2, m2, v2, mask1, y16, B.co); }
    // hnxt was consumed into y16, so writing hnxt while reading resid=hcur is safe
    launch_conv(B.co, 3, gr, stream, y16, wpack, mask1, hnxt,
                hcur, nin, B.ci, nullptr, nullptr, nullptr, nullptr, B.co, B.nin ? EPI_NIN : EPI_RES);

    float* t = hcur; hcur = hnxt; hnxt = t;
  }

  // m5 + conv5 (24->32, k5 VALID) fused with bn5-relu epilogue
  { int n = NB * S5C;
    k_m5<<<(n + 255) / 256, 256, 0, stream>>>(mask1, m5); }
  { int n = NB * 24 * S1C;
    k_cast16cl<<<(n + 255) / 256, 256, 0, stream>>>(hcur, y16, 24); }
  { long tot = 2L * (3008 / 32) * 512;
    k_pack<<<(unsigned)((tot + 255) / 256), 256, 0, stream>>>(w5, wpack, 24, 125, 3008, 32, 2); }
  {
    dim3 g5((unsigned)(NB * S5C / 16), 2);   // 25392 x 2
    launch_conv(24, 5, g5, stream, y16, wpack, m5, hnxt,
                nullptr, nullptr, 0, bn5g, bn5b, bn5m, bn5v, 32, EPI_BN5);
  }

  // Linear(FLAT->32)+ELU, then Linear(32->3)+softmax
  {
    dim3 gl((unsigned)nchunks, 4);
    k_lin1<<<gl, 256, 0, stream>>>(wl1, hnxt, part);
    k_lin1red<<<1, 64, 0, stream>>>(part, bl1, avec, nchunks);
    k_head<<<1, 32, 0, stream>>>(avec, wl2, bl2, (float*)d_out);
  }
}
